// GNNEncoder_54640573940286
// MI455X (gfx1250) — matrix operand order repaired and checked by reading, not compiled, not run on hardware
//
#include <hip/hip_runtime.h>
#include <hip/hip_bf16.h>

#define EPS 1e-5f

typedef __attribute__((ext_vector_type(16))) __bf16 v16bf;
typedef __attribute__((ext_vector_type(8)))  float  v8f;

// ---------------------------------------------------------------------------
// Degree / normalization
// ---------------------------------------------------------------------------
__global__ void deg_init_kernel(float* deg, int N) {
    int i = blockIdx.x * blockDim.x + threadIdx.x;
    if (i < N) deg[i] = 1.0f;                 // self loop contributes 1
}

__global__ void deg_edges_kernel(const int* __restrict__ ei, float* deg, long long E) {
    long long e = (long long)blockIdx.x * blockDim.x + threadIdx.x;
    if (e < E) atomicAdd(&deg[ei[E + e]], 1.0f);   // dst = ei[1][e]
}

__global__ void deg_finalize_kernel(float* deg, int N) {
    int i = blockIdx.x * blockDim.x + threadIdx.x;
    if (i < N) deg[i] = rsqrtf(fmaxf(deg[i], 1.0f));   // dis[i]
}

// ---------------------------------------------------------------------------
// Dense transform  C[N,64] = A[N,64] @ W[64,64]  via v_wmma_f32_16x16x32_bf16
// 128 threads (4 waves); each wave computes a 16x64 tile; block covers 64 rows.
// LDS staging is FRAGMENT-MAJOR: each lane reads its whole v16bf fragment as
// one aligned 32-byte LDS load (2x ds_load_b128). Full-tile blocks take a
// uniform branch-free load/store path; only the tail block predicates.
// ---------------------------------------------------------------------------
__global__ __launch_bounds__(128) void gemm64_wmma_kernel(const float* __restrict__ A,
                                                          const float* __restrict__ W,
                                                          float* __restrict__ C, int N) {
    // [frag 0..7][lane 0..31][elem 0..15], frag = tile*2 + kstep
    __shared__ __align__(32) __bf16 sAf[8 * 32 * 16];
    __shared__ __align__(32) __bf16 sWf[8 * 32 * 16];

    const int t = threadIdx.x;
    const int rowBase = blockIdx.x * 64;
    const bool fullTile = (rowBase + 64) <= N;   // uniform across block

    // --- Stage W (64x64) fragment-major: each thread handles (k,k+1) pairs.
    for (int p = t; p < 2048; p += 128) {
        const int k = (p >> 6) << 1;           // even k
        const int n = p & 63;
        const float w0 = W[k * 64 + n];
        const float w1 = W[(k + 1) * 64 + n];
        const int ks = k >> 5, kw = k & 31;
        const int g = kw >> 4;                 // lane half
        const int i = kw & 15;                 // even element index
        const int nt = n >> 4, l = n & 15;
        const int base = (((nt * 2 + ks) * 32) + (g * 16 + l)) * 16 + i;
        union { __bf16 h[2]; unsigned u; } pk;
        pk.h[0] = (__bf16)w0; pk.h[1] = (__bf16)w1;
        *(unsigned*)&sWf[base] = pk.u;         // one ds_store_b32
    }
    // --- Stage A tile fragment-major: coalesced float2 row reads.
    for (int p = t; p < 2048; p += 128) {
        const int r = p >> 5;
        const int c = (p & 31) << 1;           // even column
        const int gr = rowBase + r;
        float2 av;
        if (fullTile) {
            av = *(const float2*)(A + (long long)gr * 64 + c);
        } else {
            av = make_float2(0.f, 0.f);
            if (gr < N) av = *(const float2*)(A + (long long)gr * 64 + c);
        }
        const int ks = c >> 5, kw = c & 31;
        const int g = kw >> 4;
        const int i = kw & 15;                 // even element index
        const int w = r >> 4, l = r & 15;
        const int base = (((w * 2 + ks) * 32) + (g * 16 + l)) * 16 + i;
        union { __bf16 h[2]; unsigned u; } pk;
        pk.h[0] = (__bf16)av.x; pk.h[1] = (__bf16)av.y;
        *(unsigned*)&sAf[base] = pk.u;
    }
    __syncthreads();

    const int wave = t >> 5;
    const int lane = t & 31;
    const int l = lane & 15;
    const int g = lane >> 4;

    v8f acc[4] = {};                           // 16x64 output per wave

    #pragma unroll
    for (int ks = 0; ks < 2; ++ks) {
        const v16bf a = *(const v16bf*)&sAf[(((wave * 2 + ks) * 32) + lane) * 16];
        #pragma unroll
        for (int nt = 0; nt < 4; ++nt) {
            const v16bf b = *(const v16bf*)&sWf[(((nt * 2 + ks) * 32) + lane) * 16];
            acc[nt] = __builtin_amdgcn_wmma_f32_16x16x32_bf16(
                false, a, false, b, (short)0, acc[nt], false, false);
        }
    }

    // C/D layout: vgpr r, lane half g -> M = r + 8g ; N = l
    const long long rowStart = (long long)(rowBase + wave * 16 + 8 * g);
    float* __restrict__ cp = C + rowStart * 64 + l;
    if (fullTile) {
        // branch-free path for 1562/1563 blocks: plain coalesced b32 stores
        #pragma unroll
        for (int nt = 0; nt < 4; ++nt)
            #pragma unroll
            for (int r = 0; r < 8; ++r)
                cp[(long long)r * 64 + nt * 16] = acc[nt][r];
    } else {
        #pragma unroll
        for (int nt = 0; nt < 4; ++nt)
            #pragma unroll
            for (int r = 0; r < 8; ++r)
                if (rowStart + r < N) cp[(long long)r * 64 + nt * 16] = acc[nt][r];
    }
}

// ---------------------------------------------------------------------------
// Aggregation: h[i] = dis[i]^2 * t[i]  (self loop, also initializes h)
// then        h[dst] += dis[src]*dis[dst] * t[src]  over all edges (atomics)
// ---------------------------------------------------------------------------
__global__ void self_loop_kernel(const float* __restrict__ t, const float* __restrict__ dis,
                                 float* __restrict__ h, int N) {
    long long idx = (long long)blockIdx.x * blockDim.x + threadIdx.x;
    if (idx >= (long long)N * 64) return;
    int i = (int)(idx >> 6);
    float d = dis[i];
    h[idx] = t[idx] * d * d;
}

__global__ void edge_scatter_kernel(const int* __restrict__ ei, const float* __restrict__ dis,
                                    const float* __restrict__ t, float* __restrict__ h,
                                    long long E) {
    long long tid = (long long)blockIdx.x * blockDim.x + threadIdx.x;
    long long e = tid >> 5;                 // one wave (32 lanes) per edge
    if (e >= E) return;
    int lane = (int)(tid & 31);             // 2 features per lane
    int s = ei[e];
    int d = ei[E + e];
    float w = dis[s] * dis[d];
    const float2 tv = *(const float2*)(t + (long long)s * 64 + lane * 2);
    float* hp = h + (long long)d * 64 + lane * 2;
    atomicAdd(&hp[0], tv.x * w);
    atomicAdd(&hp[1], tv.y * w);
}

// ---------------------------------------------------------------------------
// BatchNorm: per-feature sums, then fused normalize + affine + ReLU in place
// ---------------------------------------------------------------------------
__global__ __launch_bounds__(256) void bn_stats_kernel(const float* __restrict__ h,
                                                       float* __restrict__ stats, int N) {
    const int c = blockIdx.x;               // feature column, 64 blocks
    float s = 0.f, q = 0.f;
    for (int i = threadIdx.x; i < N; i += 256) {
        float v = h[(long long)i * 64 + c];
        s += v; q += v * v;
    }
    __shared__ float ls[256], lq[256];
    ls[threadIdx.x] = s; lq[threadIdx.x] = q;
    __syncthreads();
    for (int o = 128; o > 0; o >>= 1) {
        if (threadIdx.x < o) { ls[threadIdx.x] += ls[threadIdx.x + o];
                               lq[threadIdx.x] += lq[threadIdx.x + o]; }
        __syncthreads();
    }
    if (threadIdx.x == 0) { stats[c] = ls[0]; stats[64 + c] = lq[0]; }
}

__global__ void bn_relu_kernel(float* __restrict__ h, const float* __restrict__ stats,
                               const float* __restrict__ gamma, const float* __restrict__ beta,
                               int N) {
    long long idx = (long long)blockIdx.x * blockDim.x + threadIdx.x;
    if (idx >= (long long)N * 64) return;
    int c = (int)(idx & 63);
    float inv = 1.0f / (float)N;
    float mu = stats[c] * inv;
    float var = stats[64 + c] * inv - mu * mu;       // biased (jnp.var default)
    float v = (h[idx] - mu) * rsqrtf(var + EPS) * gamma[c] + beta[c];
    h[idx] = v > 0.f ? v : 0.f;
}

// ---------------------------------------------------------------------------
// Global mean pool
// ---------------------------------------------------------------------------
__global__ void zero_kernel(float* p, long long n) {
    long long i = (long long)blockIdx.x * blockDim.x + threadIdx.x;
    if (i < n) p[i] = 0.f;
}

__global__ void pool_counts_kernel(const int* __restrict__ batch, float* counts, int N) {
    int i = blockIdx.x * blockDim.x + threadIdx.x;
    if (i < N) atomicAdd(&counts[batch[i]], 1.0f);
}

__global__ void pool_scatter_kernel(const int* __restrict__ batch, const float* __restrict__ h,
                                    float* __restrict__ out, int N) {
    long long tid = (long long)blockIdx.x * blockDim.x + threadIdx.x;
    long long i = tid >> 5;                 // one wave per node, 2 feats/lane
    if (i >= N) return;
    int lane = (int)(tid & 31);
    int gid = batch[i];
    const float2 hv = *(const float2*)(h + i * 64 + lane * 2);
    float* op = out + (long long)gid * 64 + lane * 2;
    atomicAdd(&op[0], hv.x);
    atomicAdd(&op[1], hv.y);
}

__global__ void pool_div_kernel(float* __restrict__ out, const float* __restrict__ counts, int G) {
    int idx = blockIdx.x * blockDim.x + threadIdx.x;
    if (idx >= G * 64) return;
    out[idx] /= fmaxf(counts[idx >> 6], 1.0f);
}

// ---------------------------------------------------------------------------
// Host-side orchestration
// ---------------------------------------------------------------------------
static inline int cdiv_ll(long long a, long long b) { return (int)((a + b - 1) / b); }

extern "C" void kernel_launch(void* const* d_in, const int* in_sizes, int n_in,
                              void* d_out, int out_size, void* d_ws, size_t ws_size,
                              hipStream_t stream) {
    const float* x     = (const float*)d_in[0];
    const int*   ei    = (const int*)  d_in[1];
    const int*   batch = (const int*)  d_in[2];
    const float* W1    = (const float*)d_in[3];
    const float* g1    = (const float*)d_in[5];
    const float* bt1   = (const float*)d_in[6];
    const float* W2    = (const float*)d_in[7];
    const float* g2    = (const float*)d_in[9];
    const float* bt2   = (const float*)d_in[10];
    // (b1,b2 skipped: a per-feature constant shift is exactly absorbed by BatchNorm)

    const int       N = in_sizes[0] / 64;
    const long long E = (long long)in_sizes[1] / 2;
    const int       G = out_size / 64;

    float* ws     = (float*)d_ws;
    float* dis    = ws;                               // N
    float* t      = dis + N;                          // N*64 (GEMM output)
    float* h      = t + (long long)N * 64;            // N*64 (aggregated/BN)
    float* stats  = h + (long long)N * 64;            // 128
    float* counts = stats + 128;                      // G
    float* out    = (float*)d_out;

    const int B = 256;
    const int gemmBlocks = cdiv_ll(N, 64);
    const int nfBlocks   = cdiv_ll((long long)N * 64, B);
    const int edgeBlocks = cdiv_ll(E * 32, B);
    const int nodeWBlk   = cdiv_ll((long long)N * 32, B);

    // --- normalization coefficients (once) ---
    deg_init_kernel    <<<cdiv_ll(N, B), B, 0, stream>>>(dis, N);
    deg_edges_kernel   <<<cdiv_ll(E, B), B, 0, stream>>>(ei, dis, E);
    deg_finalize_kernel<<<cdiv_ll(N, B), B, 0, stream>>>(dis, N);

    // --- layer 1: t = x@W1 ; h = propagate(t) ; h = relu(bn(h)) ---
    gemm64_wmma_kernel <<<gemmBlocks, 128, 0, stream>>>(x, W1, t, N);
    self_loop_kernel   <<<nfBlocks, B, 0, stream>>>(t, dis, h, N);
    edge_scatter_kernel<<<edgeBlocks, B, 0, stream>>>(ei, dis, t, h, E);
    bn_stats_kernel    <<<64, 256, 0, stream>>>(h, stats, N);
    bn_relu_kernel     <<<nfBlocks, B, 0, stream>>>(h, stats, g1, bt1, N);

    // --- layer 2: t = h@W2 ; h = propagate(t) ; h = relu(bn(h)) ---
    gemm64_wmma_kernel <<<gemmBlocks, 128, 0, stream>>>(h, W2, t, N);
    self_loop_kernel   <<<nfBlocks, B, 0, stream>>>(t, dis, h, N);
    edge_scatter_kernel<<<edgeBlocks, B, 0, stream>>>(ei, dis, t, h, E);
    bn_stats_kernel    <<<64, 256, 0, stream>>>(h, stats, N);
    bn_relu_kernel     <<<nfBlocks, B, 0, stream>>>(h, stats, g2, bt2, N);

    // --- global mean pool ---
    zero_kernel        <<<cdiv_ll((long long)G * 64, B), B, 0, stream>>>(out, (long long)G * 64);
    zero_kernel        <<<cdiv_ll(G, B), B, 0, stream>>>(counts, G);
    pool_counts_kernel <<<cdiv_ll(N, B), B, 0, stream>>>(batch, counts, N);
    pool_scatter_kernel<<<nodeWBlk, B, 0, stream>>>(batch, h, out, N);
    pool_div_kernel    <<<cdiv_ll((long long)G * 64, B), B, 0, stream>>>(out, counts, G);
}
